// SwitchMoe_30468497998334
// MI455X (gfx1250) — compile-verified, matrix operand order checked by reading, not run on hardware
//
#include <hip/hip_runtime.h>
#include <hip/hip_bf16.h>
#include <cstdint>

// ---------------- problem constants (from reference) ----------------
constexpr int kB = 4;          // batch
constexpr int kS = 2048;       // sequence
constexpr int kD = 2048;       // d_model
constexpr int kE = 8;          // experts
constexpr int kF = 8192;       // expert d_ff
constexpr int kC = 512;        // capacity per sequence
constexpr int kT = kB * kS;                         // tokens = 8192
constexpr int kRows = kB * (kC + 1);                // 2052 expert rows (slot C = overflow)
constexpr int kRowsPad = ((kRows + 127) / 128) * 128; // 2176 (17 tiles of 128)

typedef __attribute__((ext_vector_type(16))) __bf16 v16bf;
typedef __attribute__((ext_vector_type(8)))  float  v8f;

__device__ __forceinline__ uint16_t f32_to_bf16(float f) {
  uint32_t u = __builtin_bit_cast(uint32_t, f);
  uint32_t r = u + 0x7FFFu + ((u >> 16) & 1u);   // round-to-nearest-even
  return (uint16_t)(r >> 16);
}

union Frag16 { v16bf v; uint4 q[2]; };

// Load one 16x16 bf16 WMMA fragment row (A or Bt) from an LDS row pointer.
// Per-lane layout (CDNA5 ISA 7.12.2, 16-bit A 16x32): lane L holds row/col L&15,
// kbase = (L>>4)*8 ; elements 0..7 = K kbase..kbase+7 ; elements 8..15 = K kbase+16..+23.
__device__ __forceinline__ v16bf load_frag(const uint16_t* rowp, int kb) {
  Frag16 f;
  f.q[0] = *reinterpret_cast<const uint4*>(rowp + kb);
  f.q[1] = *reinterpret_cast<const uint4*>(rowp + kb + 16);
  return f.v;
}

// CDNA5 async global->LDS copy (16B per lane), tracked by ASYNCcnt.
// GV mode: LDS byte address taken from the vdst VGPR ; global addr = 64-bit VGPR pair.
__device__ __forceinline__ void async_copy16(void* lds_dst, const void* gsrc) {
  uint32_t l = (uint32_t)(uintptr_t)lds_dst;    // generic LDS addr low 32b = LDS byte addr
  uint64_t g = (uint64_t)(uintptr_t)gsrc;
  asm volatile("global_load_async_to_lds_b128 %0, %1, off"
               :: "v"(l), "v"(g) : "memory");
}
__device__ __forceinline__ void wait_async0() {
  asm volatile("s_wait_asynccnt 0" ::: "memory");
}

// ---------------- 1) fp32 -> bf16 transpose-convert: out[n][k] = bf16(in[k][n]) ----
__global__ __launch_bounds__(256)
void transpose_to_bf16(const float* __restrict__ in, uint16_t* __restrict__ out,
                       int K, int N, long long stride_in_e, long long stride_out_e) {
  const int e = blockIdx.z;
  const float*  ie = in  + (long long)e * stride_in_e;
  uint16_t*     oe = out + (long long)e * stride_out_e;
  __shared__ uint16_t t[32][33];
  const int tx = threadIdx.x;            // 0..31
  const int ty = threadIdx.y;            // 0..7
  const int k0 = blockIdx.y * 32;
  const int n0 = blockIdx.x * 32;
#pragma unroll
  for (int i = 0; i < 4; ++i) {
    const int k = k0 + ty + i * 8;
    t[tx][ty + i * 8] = f32_to_bf16(ie[(long long)k * N + n0 + tx]);
  }
  __syncthreads();
#pragma unroll
  for (int i = 0; i < 4; ++i) {
    const int n = n0 + ty + i * 8;
    oe[(long long)n * K + k0 + tx] = t[ty + i * 8][tx];
  }
}

// ---------------- 2) router: logits, argmax expert, top softmax prob ----------------
__global__ __launch_bounds__(256)
void router_kernel(const float* __restrict__ x, const float* __restrict__ gate_w,
                   float* __restrict__ logits_out, int* __restrict__ eidx,
                   float* __restrict__ topp) {
  const int token = blockIdx.x * 8 + (threadIdx.x >> 5);   // one wave32 per token
  const int lane  = threadIdx.x & 31;
  const float* xt = x + (long long)token * kD;
  float acc[kE];
#pragma unroll
  for (int e = 0; e < kE; ++e) acc[e] = 0.0f;
  for (int d = lane; d < kD; d += 32) {
    const float xv = xt[d];
#pragma unroll
    for (int e = 0; e < kE; ++e) acc[e] += xv * gate_w[e * kD + d];
  }
#pragma unroll
  for (int e = 0; e < kE; ++e)
    for (int off = 16; off > 0; off >>= 1) acc[e] += __shfl_xor(acc[e], off, 32);
  if (lane == 0) {
    float m = acc[0]; int am = 0;
#pragma unroll
    for (int e = 1; e < kE; ++e) if (acc[e] > m) { m = acc[e]; am = e; }
    float s = 0.0f;
#pragma unroll
    for (int e = 0; e < kE; ++e) s += __expf(acc[e] - m);
#pragma unroll
    for (int e = 0; e < kE; ++e) logits_out[(long long)token * kE + e] = acc[e];
    eidx[token] = am;
    topp[token] = 1.0f / s;   // exp(m-m)/sum == max softmax prob
  }
}

// ---------------- 3) capacity cumsum: slot per token -------------------------------
__global__ void priority_kernel(const int* __restrict__ eidx, int* __restrict__ slot) {
  const int t = threadIdx.x;
  if (t >= kB * kE) return;
  const int b = t >> 3, e = t & 7;
  const int* eb = eidx + b * kS;
  int*       sb = slot + b * kS;
  int cnt = 0;
  for (int s = 0; s < kS; ++s)
    if (eb[s] == e) { sb[s] = (cnt < kC) ? cnt : kC; ++cnt; }
}

// ---------------- 4) dispatch: scatter tokens into bf16 expert rows ----------------
__global__ __launch_bounds__(256)
void dispatch_kernel(const float* __restrict__ x, const int* __restrict__ eidx,
                     const int* __restrict__ slot, uint16_t* __restrict__ disp) {
  const int token = blockIdx.x;
  const int b = token >> 11;                 // token / kS
  const int e = eidx[token];
  const int row = b * (kC + 1) + slot[token];
  const float* src = x + (long long)token * kD;
  uint16_t* dst = disp + (long long)e * kRowsPad * kD + (long long)row * kD;
  for (int d = threadIdx.x; d < kD; d += 256) dst[d] = f32_to_bf16(src[d]);
}

// ---------------- 5) bf16 WMMA GEMM: Out = act(A[M,K] x Bt[N,K]^T) -----------------
// Block: 128x128 tile, 256 threads = 8 waves in 4x2; wave tile 32x64 = 2x4 WMMA accs.
// Double-buffered LDS fed by async global->LDS copies (ASYNCcnt), one barrier/K-step.
template <bool SILU, bool OUT_BF16>
__global__ __launch_bounds__(256)
void gemm_wmma(const uint16_t* __restrict__ A,  long long strideA_e, int lda,
               const uint16_t* __restrict__ Bt, long long strideB_e, int ldb,
               void* __restrict__ Out,          long long strideO_e, int ldo,
               int K) {
  const int e  = blockIdx.z;
  const int tm = blockIdx.y;
  const int tn = blockIdx.x;
  const uint16_t* Ae = A  + e * strideA_e + (long long)tm * 128 * lda;
  const uint16_t* Be = Bt + e * strideB_e + (long long)tn * 128 * ldb;

  __shared__ __align__(16) uint16_t As[2][128][40];   // 2 x 128x32 bf16, 80B rows
  __shared__ __align__(16) uint16_t Bs[2][128][40];

  const int tid  = threadIdx.x;
  const int wave = tid >> 5;         // 0..7
  const int lane = tid & 31;
  const int wm   = wave >> 1;        // wave row 0..3  (32 rows each)
  const int wn   = wave & 1;         // wave col 0..1  (64 cols each)
  const int rsel = lane & 15;
  const int kb   = (lane >> 4) * 8;

  // staging map: each thread owns two 16B chunks per tile (rows r and r+64)
  const int cr = tid >> 2;           // 0..63
  const int cc = (tid & 3) * 8;      // 0,8,16,24

  auto stage = [&](int buf, int k0) {
    const uint16_t* ag = Ae + (long long)cr * lda + k0 + cc;
    const uint16_t* bg = Be + (long long)cr * ldb + k0 + cc;
    async_copy16(&As[buf][cr][cc],      ag);
    async_copy16(&As[buf][cr + 64][cc], ag + 64ll * lda);
    async_copy16(&Bs[buf][cr][cc],      bg);
    async_copy16(&Bs[buf][cr + 64][cc], bg + 64ll * ldb);
  };

  v8f acc[2][4];
#pragma unroll
  for (int mi = 0; mi < 2; ++mi)
#pragma unroll
    for (int ni = 0; ni < 4; ++ni) acc[mi][ni] = 0.0f;

  stage(0, 0);                        // preload first K-slice
  int buf = 0;
  for (int k0 = 0; k0 < K; k0 += 32, buf ^= 1) {
    wait_async0();                    // current buffer's async writes complete
    __syncthreads();                  // everyone done reading other buffer + data visible
    if (k0 + 32 < K) stage(buf ^ 1, k0 + 32);   // overlap next-slice copy with WMMAs

    v16bf afr[2], bfr[4];
#pragma unroll
    for (int mi = 0; mi < 2; ++mi)
      afr[mi] = load_frag(&As[buf][wm * 32 + mi * 16 + rsel][0], kb);
#pragma unroll
    for (int ni = 0; ni < 4; ++ni)
      bfr[ni] = load_frag(&Bs[buf][wn * 64 + ni * 16 + rsel][0], kb);

#pragma unroll
    for (int mi = 0; mi < 2; ++mi)
#pragma unroll
      for (int ni = 0; ni < 4; ++ni)
        acc[mi][ni] = __builtin_amdgcn_wmma_f32_16x16x32_bf16(
            false, afr[mi], false, bfr[ni], (short)0, acc[mi][ni], false, false);
  }

  // C/D layout: element r <-> M = r + 8*(lane>>4), N = lane&15
  const int colbase = tn * 128 + wn * 64 + rsel;
  const int rowadd  = (lane >> 4) * 8;
#pragma unroll
  for (int mi = 0; mi < 2; ++mi) {
#pragma unroll
    for (int ni = 0; ni < 4; ++ni) {
      const int colg = colbase + ni * 16;
#pragma unroll
      for (int r = 0; r < 8; ++r) {
        const int rowg = tm * 128 + wm * 32 + mi * 16 + rowadd + r;
        float v = acc[mi][ni][r];
        if (SILU) v = v / (1.0f + __expf(-v));
        const long long idx = e * strideO_e + (long long)rowg * ldo + colg;
        if (OUT_BF16) reinterpret_cast<uint16_t*>(Out)[idx] = f32_to_bf16(v);
        else          reinterpret_cast<float*>(Out)[idx] = v;
      }
    }
  }
}

// ---------------- 6) combine: gather + prob scale + expert index -------------------
__global__ __launch_bounds__(256)
void combine_kernel(const float* __restrict__ x, const float* __restrict__ eout,
                    const int* __restrict__ eidx, const int* __restrict__ slot,
                    const float* __restrict__ topp, float* __restrict__ out) {
  const int token = blockIdx.x;
  const int b  = token >> 11;
  const int e  = eidx[token];
  const int sl = slot[token];
  const bool routed = sl < kC;
  const float tp = topp[token];
  const int row = b * (kC + 1) + sl;
  const float* src = routed ? (eout + (long long)e * kRowsPad * kD + (long long)row * kD)
                            : (x + (long long)token * kD);
  float* dst = out + (long long)token * kD;
  for (int d = threadIdx.x; d < kD; d += 256) dst[d] = tp * src[d];
  if (threadIdx.x == 0)
    out[(long long)kT * kD + (long long)kT * kE + token] = (float)(routed ? e : 0);
}

// ---------------- host launcher ----------------------------------------------------
extern "C" void kernel_launch(void* const* d_in, const int* in_sizes, int n_in,
                              void* d_out, int out_size, void* d_ws, size_t ws_size,
                              hipStream_t stream) {
  (void)in_sizes; (void)n_in; (void)out_size; (void)ws_size;
  const float* x      = (const float*)d_in[0];
  const float* gate_w = (const float*)d_in[1];
  const float* w1     = (const float*)d_in[2];
  const float* w2     = (const float*)d_in[3];

  float* out    = (float*)d_out;
  float* logits = out + (size_t)kT * kD;

  // workspace layout (~0.89 GB); eout aliases w1t (dead after GEMM1)
  char* ws = (char*)d_ws;
  size_t off = 0;
  int*   eidx = (int*)(ws + off);   off += (size_t)kT * 4;
  int*   slot = (int*)(ws + off);   off += (size_t)kT * 4;
  float* topp = (float*)(ws + off); off += (size_t)kT * 4;
  off = (off + 255) & ~(size_t)255;
  uint16_t* disp = (uint16_t*)(ws + off); off += (size_t)kE * kRowsPad * kD * 2;
  const size_t off_w1t = off;
  uint16_t* w1t  = (uint16_t*)(ws + off); off += (size_t)kE * kF * kD * 2;
  uint16_t* w2t  = (uint16_t*)(ws + off); off += (size_t)kE * kD * kF * 2;
  uint16_t* h    = (uint16_t*)(ws + off); off += (size_t)kE * kRowsPad * kF * 2;
  float*    eout = (float*)(ws + off_w1t);

  dim3 tb(32, 8);
  // w1[e]: [kD][kF] -> w1t[e]: [kF][kD]
  transpose_to_bf16<<<dim3(kF / 32, kD / 32, kE), tb, 0, stream>>>(
      w1, w1t, kD, kF, (long long)kD * kF, (long long)kF * kD);
  // w2[e]: [kF][kD] -> w2t[e]: [kD][kF]
  transpose_to_bf16<<<dim3(kD / 32, kF / 32, kE), tb, 0, stream>>>(
      w2, w2t, kF, kD, (long long)kF * kD, (long long)kD * kF);

  router_kernel<<<kT / 8, 256, 0, stream>>>(x, gate_w, logits, eidx, topp);
  priority_kernel<<<1, 32, 0, stream>>>(eidx, slot);
  dispatch_kernel<<<kT, 256, 0, stream>>>(x, eidx, slot, disp);

  // GEMM1: h = silu(disp x w1), bf16 out. A:[2176,2048] Bt:[8192,2048] K=2048
  gemm_wmma<true, true><<<dim3(kF / 128, kRowsPad / 128, kE), 256, 0, stream>>>(
      disp, (long long)kRowsPad * kD, kD,
      w1t,  (long long)kF * kD,       kD,
      (void*)h, (long long)kRowsPad * kF, kF, kD);

  // GEMM2: eout = h x w2, f32 out. A:[2176,8192] Bt:[2048,8192] K=8192
  gemm_wmma<false, false><<<dim3(kD / 128, kRowsPad / 128, kE), 256, 0, stream>>>(
      h,    (long long)kRowsPad * kF, kF,
      w2t,  (long long)kD * kF,       kF,
      (void*)eout, (long long)kRowsPad * kD, kD, kF);

  combine_kernel<<<kT, 256, 0, stream>>>(x, eout, eidx, slot, topp, out);
}
